// Attention_60790967107825
// MI455X (gfx1250) — matrix operand order repaired and checked by reading, not compiled, not run on hardware
//
#include <hip/hip_runtime.h>
#include <hip/hip_bf16.h>
#include <math.h>

typedef __attribute__((ext_vector_type(16))) __bf16 v16bf;
typedef __attribute__((ext_vector_type(8)))  __bf16 v8bf;
typedef __attribute__((ext_vector_type(8)))  float  v8f;
typedef __attribute__((ext_vector_type(4)))  unsigned int v4u;
typedef __attribute__((ext_vector_type(8)))  int    v8i;
typedef __attribute__((ext_vector_type(4)))  int    v4i;

#define B_    4
#define H_    16
#define N_    2048
#define D_    64
#define DIM_  1024
#define SCALE_ 0.125f  /* 64^-0.5 */

#if defined(__has_builtin)
#if __has_builtin(__builtin_amdgcn_tensor_load_to_lds) && __has_builtin(__builtin_amdgcn_s_wait_tensorcnt)
#define HAVE_TDM 1
#endif
#endif

// ---------------------------------------------------------------------------
// fp32 -> bf16 copy (grid-stride)
// ---------------------------------------------------------------------------
__global__ void cvt_kernel(const float* __restrict__ src, __bf16* __restrict__ dst, int n)
{
    for (int i = blockIdx.x * blockDim.x + threadIdx.x; i < n; i += gridDim.x * blockDim.x)
        dst[i] = (__bf16)src[i];
}

// fp32 [BH, N, D] -> bf16 transposed [BH, D, N]
__global__ void cvt_vt_kernel(const float* __restrict__ v, __bf16* __restrict__ vt)
{
    int i    = blockIdx.x * blockDim.x + threadIdx.x;   // over B*H*D*N
    int n    = i & (N_ - 1);
    int rest = i >> 11;                                 // / N_
    int d    = rest & (D_ - 1);
    int bh   = rest >> 6;                               // / D_
    vt[i] = (__bf16)v[((size_t)bh * N_ + n) * D_ + d];
}

#ifdef HAVE_TDM
// ---------------------------------------------------------------------------
// TDM: DMA a 32-row x 64-col bf16 K tile (row stride 64 elems) into LDS.
// D# packing per CDNA5 ISA.
// 6-arg toolchain form: (g0 u32x4, g1 i32x8, g2 i32x4, g3 i32x4, i32x8, cpol)
// ---------------------------------------------------------------------------
__device__ inline void tdm_load_k_tile(const __bf16* gsrc, unsigned lds_off)
{
    unsigned long long ga = (unsigned long long)(uintptr_t)gsrc;
    v4u g0;
    g0[0] = 1u;                                            // count=1, user mode
    g0[1] = lds_off;                                       // LDS byte address
    g0[2] = (unsigned)(ga & 0xFFFFFFFFu);                  // global_addr lo
    g0[3] = (unsigned)((ga >> 32) & 0x01FFFFFFu) | (2u << 30);  // addr hi | type=2
    v8i g1;
    g1[0] = (1 << 16);                                     // data_size=1 (2 bytes)
    g1[1] = (int)(64u << 16);                              // tensor_dim0[15:0]=64
    g1[2] = (int)(2048u << 16);                            // dim0 hi=0 | tensor_dim1 lo=2048
    g1[3] = (int)(64u << 16);                              // dim1 hi=0 | tile_dim0=64
    g1[4] = 32;                                            // tile_dim1=32, tile_dim2=0
    g1[5] = 64;                                            // tensor_dim0_stride=64
    g1[6] = 0;
    g1[7] = 0;
    v4i z4 = {0, 0, 0, 0};
    v8i z8 = {0, 0, 0, 0, 0, 0, 0, 0};
    __builtin_amdgcn_tensor_load_to_lds(g0, g1, z4, z4, z8, 0);
}
#endif

// ---------------------------------------------------------------------------
// Flash-attention: one wave per 16 query rows, 4 waves/block (64 rows).
// K tiles staged in LDS by the TDM (double buffered); V pre-transposed bf16.
// ---------------------------------------------------------------------------
__global__ __launch_bounds__(128) void attn_kernel(
    const float*  __restrict__ q,    const __bf16* __restrict__ kbf,
    const __bf16* __restrict__ vt,   const unsigned char* __restrict__ mask,
    __bf16* __restrict__ xout)
{
#ifdef HAVE_TDM
    __shared__ __attribute__((aligned(128))) __bf16 lds_k[2][32 * 64];
#endif
    __shared__ __attribute__((aligned(32)))  __bf16 lds_p[4][16 * 32];

    const int lane  = threadIdx.x & 31;
    const int wave  = threadIdx.x >> 5;
    const int ntile = N_ / 64;
    const int bh    = blockIdx.x / ntile;
    const int b     = bh / H_;
    const int h     = bh % H_;
    const int qbase = (blockIdx.x % ntile) * 64 + wave * 16;

    const int col    = lane & 15;
    const int hi     = lane >> 4;
    const int kbase8 = hi * 8;

    const float*  qp  = q   + (size_t)bh * N_ * D_;
    const __bf16* kbh = kbf + (size_t)bh * N_ * D_;
    const __bf16* vbh = vt  + (size_t)bh * D_ * N_;

    // ---- Q as two A-fragments ----
    v16bf qa[2];
    {
        const float* qrow = qp + (size_t)(qbase + col) * D_;
        for (int f = 0; f < 2; ++f)
            for (int e = 0; e < 16; ++e)
                qa[f][e] = (__bf16)qrow[f * 32 + hi * 16 + e];
    }

    // ---- hoisted masks: rows (8 bools) + per-lane column bitset (128 bits) ----
    bool rowm[8];
    for (int r = 0; r < 8; ++r) {
        int row = qbase + r + hi * 8;
        int idx = (row == 0) ? 0 : row - 1;
        rowm[r] = (row == 0) | (mask[b * (N_ - 1) + idx] != 0);
    }
    unsigned cbits[4] = {0u, 0u, 0u, 0u};
    for (int i = 0; i < 128; ++i) {
        int key = i * 16 + col;
        int idx = (key == 0) ? 0 : key - 1;
        bool mm = (key == 0) | (mask[b * (N_ - 1) + idx] != 0);
        cbits[i >> 5] |= ((unsigned)mm) << (i & 31);
    }

    float mrow[8], lrow[8];
    v8f   o[4];
    for (int r = 0; r < 8; ++r) { mrow[r] = -3.0e38f; lrow[r] = 0.f; }
    for (int t = 0; t < 4; ++t)
        for (int r = 0; r < 8; ++r) o[t][r] = 0.f;

#ifdef HAVE_TDM
    if (wave == 0)
        tdm_load_k_tile(kbh, (unsigned)(uintptr_t)&lds_k[0][0]);
#endif

    for (int it = 0; it < N_ / 32; ++it) {
        const int j = it * 32;
#ifdef HAVE_TDM
        if (wave == 0) __builtin_amdgcn_s_wait_tensorcnt(0);
        __syncthreads();                       // K tile (it&1) ready, prev buf free
        if (wave == 0 && it + 1 < N_ / 32)
            tdm_load_k_tile(kbh + (size_t)(j + 32) * D_,
                            (unsigned)(uintptr_t)&lds_k[(it + 1) & 1][0]);
        const __bf16* ktile = lds_k[it & 1];
#endif
        if (j + 32 < N_)
            __builtin_prefetch(vbh + (size_t)col * N_ + j + 32, 0, 1);

        // ---- S = Q K^T : two 16x16 tiles ----
        v8f s[2];
        for (int jt = 0; jt < 2; ++jt) {
            v8f acc = {};
            for (int f = 0; f < 2; ++f) {
#ifdef HAVE_TDM
                v16bf kb = *(const v16bf*)&ktile[(jt * 16 + col) * 64 + f * 32 + hi * 16];
#else
                v16bf kb = *(const v16bf*)&kbh[(size_t)(j + jt * 16 + col) * D_ + f * 32 + hi * 16];
#endif
                acc = __builtin_amdgcn_wmma_f32_16x16x32_bf16(
                          false, qa[f], false, kb, (short)0, acc, false, false);
            }
            s[jt] = acc;
        }

        // ---- scale + mask (bit tests, no loads) ----
        const int i0 = it * 2, i1 = it * 2 + 1;
        const bool c0 = (cbits[i0 >> 5] >> (i0 & 31)) & 1u;
        const bool c1 = (cbits[i1 >> 5] >> (i1 & 31)) & 1u;
        for (int r = 0; r < 8; ++r) {
            s[0][r] = (rowm[r] && c0) ? s[0][r] * SCALE_ : -1.0e30f;
            s[1][r] = (rowm[r] && c1) ? s[1][r] * SCALE_ : -1.0e30f;
        }

        // ---- online softmax (row spans 16 lanes in C layout) ----
        float p0[8], p1[8];
        for (int r = 0; r < 8; ++r) {
            float t = fmaxf(s[0][r], s[1][r]);
            for (int x = 1; x < 16; x <<= 1) t = fmaxf(t, __shfl_xor(t, x, 16));
            float mnew  = fmaxf(mrow[r], t);
            float alpha = __expf(mrow[r] - mnew);
            mrow[r] = mnew;
            p0[r] = __expf(s[0][r] - mnew);
            p1[r] = __expf(s[1][r] - mnew);
            float rs = p0[r] + p1[r];
            for (int x = 1; x < 16; x <<= 1) rs += __shfl_xor(rs, x, 16);
            lrow[r] = lrow[r] * alpha + rs;
            for (int t4 = 0; t4 < 4; ++t4) o[t4][r] *= alpha;
        }

        // ---- P: C layout -> A layout through LDS ----
        __bf16* pl = lds_p[wave];
        for (int r = 0; r < 8; ++r) {
            pl[(r + hi * 8) * 32 + col]      = (__bf16)p0[r];
            pl[(r + hi * 8) * 32 + 16 + col] = (__bf16)p1[r];
        }
        __syncthreads();
        v16bf pa;
        {
            v8bf lo  = *(const v8bf*)&pl[col * 32 + hi * 16];
            v8bf hi8 = *(const v8bf*)&pl[col * 32 + hi * 16 + 8];
            for (int e = 0; e < 8; ++e) { pa[e] = lo[e]; pa[e + 8] = hi8[e]; }
        }
        __syncthreads();

        // ---- O += P V : V^T pre-staged bf16, contiguous B-fragments ----
        for (int t = 0; t < 4; ++t) {
            v16bf vb = *(const v16bf*)&vbh[(size_t)(t * 16 + col) * N_ + j + hi * 16];
            o[t] = __builtin_amdgcn_wmma_f32_16x16x32_bf16(
                       false, pa, false, vb, (short)0, o[t], false, false);
        }
    }

    // ---- finalize: O/l, write bf16 X transposed [B, N, H*D] ----
    for (int r = 0; r < 8; ++r) {
        float inv = 1.0f / lrow[r];
        int   row = qbase + r + hi * 8;
        __bf16* orow = xout + (size_t)(b * N_ + row) * DIM_ + h * D_;
        for (int t = 0; t < 4; ++t)
            orow[t * 16 + col] = (__bf16)(o[t][r] * inv);
    }
}

// ---------------------------------------------------------------------------
// Projection: out[m][n] = sum_k X[m][k] * W[n][k] + bias[n]  (bf16 WMMA)
// ---------------------------------------------------------------------------
__global__ __launch_bounds__(128) void proj_kernel(
    const __bf16* __restrict__ X, const __bf16* __restrict__ W,
    const float* __restrict__ bias, float* __restrict__ out)
{
    const int lane   = threadIdx.x & 31;
    const int wave   = threadIdx.x >> 5;
    const int col    = lane & 15;
    const int hi     = lane >> 4;
    const int kbase8 = hi * 8;

    const int mtiles = (B_ * N_) / 64;
    const int mbase  = (blockIdx.x % mtiles) * 64 + wave * 16;
    const int nbase  = (blockIdx.x / mtiles) * 64;

    v8f acc[4];
    for (int t = 0; t < 4; ++t)
        for (int r = 0; r < 8; ++r) acc[t][r] = 0.f;

    const __bf16* xrow = X + (size_t)(mbase + col) * DIM_;
    for (int kk = 0; kk < DIM_; kk += 32) {
        v16bf xa;
        {
            v8bf lo  = *(const v8bf*)&xrow[kk + hi * 16];
            v8bf hi8 = *(const v8bf*)&xrow[kk + hi * 16 + 8];
            for (int e = 0; e < 8; ++e) { xa[e] = lo[e]; xa[e + 8] = hi8[e]; }
        }
        for (int t = 0; t < 4; ++t) {
            v16bf wb = *(const v16bf*)&W[(size_t)(nbase + t * 16 + col) * DIM_ + kk + hi * 16];
            acc[t] = __builtin_amdgcn_wmma_f32_16x16x32_bf16(
                         false, xa, false, wb, (short)0, acc[t], false, false);
        }
    }

    for (int r = 0; r < 8; ++r) {
        int    row  = mbase + r + hi * 8;
        float* orow = out + (size_t)row * DIM_;
        for (int t = 0; t < 4; ++t) {
            int n = nbase + t * 16 + col;
            orow[n] = acc[t][r] + bias[n];
        }
    }
}

// ---------------------------------------------------------------------------
extern "C" void kernel_launch(void* const* d_in, const int* in_sizes, int n_in,
                              void* d_out, int out_size, void* d_ws, size_t ws_size,
                              hipStream_t stream)
{
    const float*         q    = (const float*)d_in[0];
    const float*         k    = (const float*)d_in[1];
    const float*         v    = (const float*)d_in[2];
    const unsigned char* mask = (const unsigned char*)d_in[3];
    const float*         W    = (const float*)d_in[4];
    const float*         bias = (const float*)d_in[5];
    float*               out  = (float*)d_out;

    __bf16* wsb = (__bf16*)d_ws;
    __bf16* kbf = wsb;                               // [BH, N, D]   8388608
    __bf16* vt  = wsb + 8388608;                     // [BH, D, N]   8388608
    __bf16* xbf = wsb + 16777216;                    // [B*N, DIM]   8388608
    __bf16* wbf = wsb + 25165824;                    // [DIM, DIM]   1048576

    const int KN = B_ * H_ * N_ * D_;
    const int WN = DIM_ * DIM_;
    cvt_kernel   <<<2048, 256, 0, stream>>>(k, kbf, KN);
    cvt_vt_kernel<<<KN / 256, 256, 0, stream>>>(v, vt);
    cvt_kernel   <<<1024, 256, 0, stream>>>(W, wbf, WN);

    attn_kernel<<<B_ * H_ * (N_ / 64), 128, 0, stream>>>(q, kbf, vt, mask, xbf);
    proj_kernel<<<((B_ * N_) / 64) * (DIM_ / 64), 128, 0, stream>>>(xbf, wbf, bias, out);
}
